// SkipGram_13709535608898
// MI455X (gfx1250) — matrix operand order repaired and checked by reading, not compiled, not run on hardware
//
#include <hip/hip_runtime.h>
#include <math.h>
#include <stdint.h>

// ---------------------------------------------------------------------------
// Problem constants (from reference): B=1024, K=5, VOC=32000, D=128
// ---------------------------------------------------------------------------
#define BATCH   1024
#define KNEG    5
#define VOC     32000
#define DIM     128
#define KT_TOT  (VOC / 32)            // 1000 K-tiles of 32
#define KSPLIT  5
#define KT_PER  (KT_TOT / KSPLIT)     // 200
#define STAGE_KT 4                    // K-tiles per async stage
#define NSTAGE  (KT_PER / STAGE_KT)   // 50
#define ROWS    (BATCH + BATCH + BATCH * KNEG)  // 7168 GEMM rows total
#define CT      (DIM / 16)            // 8 column tiles
#define BCHUNK  (CT * 32 * 32)        // bytes of packed B per K-tile = 8192
#define STAGE_BYTES (STAGE_KT * BCHUNK)         // 32768

typedef __attribute__((ext_vector_type(16))) __bf16 v16bf;
typedef __attribute__((ext_vector_type(8)))  float  v8f;
typedef __attribute__((ext_vector_type(4)))  float  flt4;

// Workspace layout (bytes)
#define VP_BYTES ((size_t)KT_TOT * CT * 32 * 16 * 2)            // 8,192,000
#define EP_BYTES ((size_t)KSPLIT * ROWS * DIM * 4)              // 18,350,080

// ---------------------------------------------------------------------------
// Kernel 1: repack f32 weight matrix [VOC, 128] into bf16 WMMA B-fragment
// layout: Wp[(kt*8 + ctile)*32 + lane][16 bf16]; lane 0..15 = column
// ctile*16 + lane with K = kt*32 + 0..15, lane 16..31 = same columns, K + 16.
// Contiguous k order, same as the GEMM's A -> GEMM reads 32B contiguous per lane.
// ---------------------------------------------------------------------------
__global__ __launch_bounds__(256) void pack_bf16(const float* __restrict__ W,
                                                 __bf16* __restrict__ Wp) {
  int idx = blockIdx.x * 256 + threadIdx.x;     // 0 .. 1000*8*32 - 1
  int t   = idx >> 8;                           // K-tile
  int rem = idx & 255;
  int c   = rem >> 5;                           // column tile
  int l   = rem & 31;                           // lane
  int col  = c * 16 + (l & 15);
  int koff = (l < 16) ? 0 : 16;
  v16bf out;
#pragma unroll
  for (int e = 0; e < 16; ++e)
    out[e] = (__bf16)W[(size_t)(t * 32 + koff + e) * DIM + col];
  *(v16bf*)(Wp + (size_t)idx * 16) = out;
}

// ---------------------------------------------------------------------------
// Kernel 2: bf16 WMMA GEMM, async-to-LDS double-buffered B staging,
// 4 K-tiles (32KB) per stage -> 64 straight-line WMMAs between barriers.
// Block = 128 threads (4 waves). Each wave: 32 rows x 128 cols (16 WMMA tiles).
// gridDim.x = ROWS/128 (56), gridDim.y = KSPLIT (5).
// Rows 0..1023 = vi (x Vp), 1024..2047 = vo (x Up), 2048..7167 = neg (x Up).
// ---------------------------------------------------------------------------
__global__ __launch_bounds__(128) void gemm_embed(
    const float* __restrict__ vi, const float* __restrict__ vo,
    const float* __restrict__ neg, const __bf16* __restrict__ Vp,
    const __bf16* __restrict__ Up, float* __restrict__ Ep) {
  __shared__ __align__(16) __bf16 Bs[2][STAGE_KT * CT * 32 * 16];  // 2 x 32 KB

  const int lane = threadIdx.x & 31;
  const int wave = threadIdx.x >> 5;
  const int m0   = blockIdx.x * 128 + wave * 32;   // first row of this wave
  const int l15  = lane & 15;
  const int hi   = lane >> 4;                      // 0: low half-wave, 1: high

  // A-row pointers for the wave's two 16-row fragments
  const float* arow[2];
#pragma unroll
  for (int r = 0; r < 2; ++r) {
    int rowA = m0 + r * 16 + l15;
    if (rowA < BATCH)           arow[r] = vi  + (size_t)rowA * VOC;
    else if (rowA < 2 * BATCH)  arow[r] = vo  + (size_t)(rowA - BATCH) * VOC;
    else                        arow[r] = neg + (size_t)(rowA - 2 * BATCH) * VOC;
  }
  const __bf16* Bp = (m0 < BATCH) ? Vp : Up;   // 128-row block stays in one segment

  const int kt0 = blockIdx.y * KT_PER;

  // LDS byte offset of Bs (generic LDS pointer: low 32 bits = LDS offset)
  const uint32_t ldsBase = (uint32_t)(uintptr_t)(&Bs[0][0]);

  // Stage STAGE_KT packed-B K-tiles (32 KB) into LDS buffer `buf`.
  // 128 threads x 256B each; inst offset applies to BOTH lds and mem address.
#define ALDS(off) "global_load_async_to_lds_b128 %0, %1, off offset:" #off "\n\t"
  auto stage = [&](int buf, int ktFirst) {
    const char* src = (const char*)Bp + (size_t)ktFirst * BCHUNK
                      + (size_t)threadIdx.x * 256;
    uint32_t    dst = ldsBase + (uint32_t)buf * STAGE_BYTES
                      + (uint32_t)threadIdx.x * 256;
    asm volatile(
        ALDS(0)   ALDS(16)  ALDS(32)  ALDS(48)
        ALDS(64)  ALDS(80)  ALDS(96)  ALDS(112)
        ALDS(128) ALDS(144) ALDS(160) ALDS(176)
        ALDS(192) ALDS(208) ALDS(224) "global_load_async_to_lds_b128 %0, %1, off offset:240"
        :: "v"(dst), "v"(src)
        : "memory");
  };
#undef ALDS

  v8f acc[2][CT] = {};

  stage(0, kt0);

  for (int is = 0; is < NSTAGE; ++is) {
    const int cur = is & 1;

    // Wait for my async stage, then make it visible workgroup-wide.
    asm volatile("s_wait_asynccnt 0x0" ::: "memory");
    __syncthreads();

    // Prefetch next 32KB into the other buffer while we compute.
    if (is + 1 < NSTAGE) stage(cur ^ 1, kt0 + (is + 1) * STAGE_KT);

    // 64 WMMAs of straight-line code between barriers.
#pragma unroll
    for (int kl = 0; kl < STAGE_KT; ++kl) {
      const int kt = kt0 + is * STAGE_KT + kl;

      // --- A fragments (streamed once: non-temporal, keep B hot in L2).
      // Lanes 0-15 hold K {0..15}; lanes 16-31 hold K {16..31}.
      v16bf af[2];
#pragma unroll
      for (int r = 0; r < 2; ++r) {
        const float* ap = arow[r] + kt * 32 + hi * 16;
        flt4 a0 = __builtin_nontemporal_load((const flt4*)(ap));
        flt4 a1 = __builtin_nontemporal_load((const flt4*)(ap + 4));
        flt4 a2 = __builtin_nontemporal_load((const flt4*)(ap + 8));
        flt4 a3 = __builtin_nontemporal_load((const flt4*)(ap + 12));
#pragma unroll
        for (int j = 0; j < 4; ++j) {
          af[r][j]      = (__bf16)a0[j];   // K = kbase + 0..3
          af[r][4 + j]  = (__bf16)a1[j];   // K = kbase + 4..7
          af[r][8 + j]  = (__bf16)a2[j];   // K = kbase + 8..11
          af[r][12 + j] = (__bf16)a3[j];   // K = kbase + 12..15
        }
      }

      // --- B fragments from LDS, 2 WMMAs per fragment
#pragma unroll
      for (int c = 0; c < CT; ++c) {
        v16bf bf = *(const v16bf*)(&Bs[cur][((kl * CT + c) * 32 + lane) * 16]);
        acc[0][c] = __builtin_amdgcn_wmma_f32_16x16x32_bf16(
            false, af[0], false, bf, (short)0, acc[0][c], false, false);
        acc[1][c] = __builtin_amdgcn_wmma_f32_16x16x32_bf16(
            false, af[1], false, bf, (short)0, acc[1][c], false, false);
      }
    }

    // All waves done reading Bs[cur] before it gets overwritten next stage.
    __syncthreads();
  }

  // --- store partial results (C/D layout: VGPR v -> row +v (+8 for hi lanes),
  //     lane&15 -> column within the 16-wide tile)
  float* ep = Ep + (size_t)blockIdx.y * ROWS * DIM;
#pragma unroll
  for (int r = 0; r < 2; ++r) {
#pragma unroll
    for (int c = 0; c < CT; ++c) {
      int col = c * 16 + l15;
#pragma unroll
      for (int v = 0; v < 8; ++v) {
        int row = m0 + r * 16 + v + hi * 8;
        ep[(size_t)row * DIM + col] = acc[r][c][v];
      }
    }
  }
}

// ---------------------------------------------------------------------------
// Kernel 3: per-batch epilogue. Fold KSPLIT partials, compute the 6 dots,
// logsigmoid terms, write per-row loss. One 128-thread block per batch elem.
// ---------------------------------------------------------------------------
__global__ __launch_bounds__(128) void epilogue(const float* __restrict__ Ep,
                                                float* __restrict__ loss) {
  const int b = blockIdx.x;
  const int d = threadIdx.x;

  auto sumRow = [&](int row) {
    float s = 0.f;
#pragma unroll
    for (int p = 0; p < KSPLIT; ++p)
      s += Ep[(size_t)p * ROWS * DIM + (size_t)row * DIM + d];
    return s;
  };

  float vie = sumRow(b);
  float voe = sumRow(BATCH + b);

  __shared__ float red[6][DIM];
  red[0][d] = vie * voe;
#pragma unroll
  for (int k = 0; k < KNEG; ++k)
    red[1 + k][d] = vie * sumRow(2 * BATCH + b * KNEG + k);
  __syncthreads();

  for (int s = 64; s > 0; s >>= 1) {
    if (d < s) {
      for (int j = 0; j < 6; ++j) red[j][d] += red[j][d + s];
    }
    __syncthreads();
  }

  if (d == 0) {
    auto ls = [](float x) {  // numerically stable log_sigmoid
      return fminf(x, 0.f) - log1pf(expf(-fabsf(x)));
    };
    float l = -ls(red[0][0]);
    for (int k = 0; k < KNEG; ++k) l -= ls(-red[1 + k][0]);
    loss[b] = l;
  }
}

// ---------------------------------------------------------------------------
// Kernel 4: deterministic reduce of 1024 losses -> mean scalar.
// ---------------------------------------------------------------------------
__global__ __launch_bounds__(256) void reduce_loss(const float* __restrict__ loss,
                                                   float* __restrict__ out) {
  __shared__ float sm[256];
  float s = 0.f;
  for (int i = threadIdx.x; i < BATCH; i += 256) s += loss[i];
  sm[threadIdx.x] = s;
  __syncthreads();
  for (int st = 128; st > 0; st >>= 1) {
    if (threadIdx.x < st) sm[threadIdx.x] += sm[threadIdx.x + st];
    __syncthreads();
  }
  if (threadIdx.x == 0) out[0] = sm[0] / (float)BATCH;
}

// ---------------------------------------------------------------------------
extern "C" void kernel_launch(void* const* d_in, const int* in_sizes, int n_in,
                              void* d_out, int out_size, void* d_ws, size_t ws_size,
                              hipStream_t stream) {
  const float* vi  = (const float*)d_in[0];
  const float* vo  = (const float*)d_in[1];
  const float* neg = (const float*)d_in[2];
  const float* V   = (const float*)d_in[3];
  const float* U   = (const float*)d_in[4];

  char* ws = (char*)d_ws;
  __bf16* Vp  = (__bf16*)(ws);
  __bf16* Up  = (__bf16*)(ws + VP_BYTES);
  float*  Ep  = (float*)(ws + 2 * VP_BYTES);
  float*  lpb = (float*)(ws + 2 * VP_BYTES + EP_BYTES);

  // 1) repack weights into bf16 WMMA-fragment layout (L2-resident afterwards)
  pack_bf16<<<KT_TOT, 256, 0, stream>>>(V, Vp);
  pack_bf16<<<KT_TOT, 256, 0, stream>>>(U, Up);

  // 2) big bf16 WMMA GEMM, K split 5-ways, 32KB async-to-LDS staging
  dim3 grid(ROWS / 128, KSPLIT);
  gemm_embed<<<grid, 128, 0, stream>>>(vi, vo, neg, Vp, Up, Ep);

  // 3) per-batch dots + logsigmoid
  epilogue<<<BATCH, 128, 0, stream>>>(Ep, lpb);

  // 4) mean
  reduce_loss<<<1, 256, 0, stream>>>(lpb, (float*)d_out);
}
